// SplineCNN8Residuals_6828998001345
// MI455X (gfx1250) — matrix operand order repaired and checked by reading, not compiled, not run on hardware
//
#include <hip/hip_runtime.h>
#include <math.h>

// ---------------- problem constants ----------------
#define NNODES 10000
#define NEDGES 160000
#define SKNOTS 27          // 3^3 spline slots
#define BN_EPS 1e-5f
#define NROWT (NNODES / 16)   // 625 row tiles
#define FINP_MAX 288          // max padded K (fin=259 -> 288)

typedef __attribute__((ext_vector_type(16))) _Float16 v16h;
typedef __attribute__((ext_vector_type(8)))  float    v8f;

static inline int cdiv(int a, int b) { return (a + b - 1) / b; }

// ---------------- elementwise / setup kernels ----------------

__global__ void k_zero(float* p, int n) {
  int i = blockIdx.x * blockDim.x + threadIdx.x;
  if (i < n) p[i] = 0.f;
}

// per-edge trilinear basis weights + weight-slot indices (shared by all 24 convs)
__global__ void k_edge_basis(const float* __restrict__ ea,
                             float* __restrict__ ebas, int* __restrict__ ewi) {
  int e = blockIdx.x * blockDim.x + threadIdx.x;
  if (e >= NEDGES) return;
  float fr[3]; int kv[3];
#pragma unroll
  for (int d = 0; d < 3; ++d) {
    float p = ea[e * 3 + d] * 3.f;   // K=3
    float k = floorf(p);
    kv[d] = (int)k;
    fr[d] = p - k;
  }
#pragma unroll
  for (int c = 0; c < 8; ++c) {
    float bas = 1.f; int wi = 0; int pw = 1;
#pragma unroll
    for (int d = 0; d < 3; ++d) {
      int bit = (c >> d) & 1;
      bas *= bit ? fr[d] : (1.f - fr[d]);
      wi += ((kv[d] + bit) % 3) * pw;
      pw *= 3;
    }
    ebas[e * 8 + c] = bas;
    ewi[e * 8 + c] = wi;
  }
}

__global__ void k_deg(const int* __restrict__ ei, float* __restrict__ deg) {
  int e = blockIdx.x * blockDim.x + threadIdx.x;
  if (e < NEDGES) atomicAdd(&deg[ei[NEDGES + e]], 1.f);
}

__global__ void k_deg_inv(float* deg) {
  int n = blockIdx.x * blockDim.x + threadIdx.x;
  if (n < NNODES) deg[n] = 1.f / fmaxf(deg[n], 1.f);
}

// Xh[n][finp] = f16(X[n][fin]) zero-padded to finp (multiple of 32)
__global__ void k_pack_x(const float* __restrict__ X, _Float16* __restrict__ Xh,
                         int fin, int finp) {
  int idx = blockIdx.x * blockDim.x + threadIdx.x;
  if (idx >= NNODES * finp) return;
  int n = idx / finp, k = idx % finp;
  Xh[idx] = (k < fin) ? (_Float16)X[n * fin + k] : (_Float16)0.f;
}

// WtT[col][finp] (K-major, f16, zero-padded): cols 0..27*fc-1 = W[s,:,o0+o],
// cols 27*fc..28*fc-1 = root[:,o0+o]
__global__ void k_pack_w(const float* __restrict__ W, const float* __restrict__ root,
                         _Float16* __restrict__ WtT, int fin, int finp,
                         int fout, int fc, int o0) {
  int idx = blockIdx.x * blockDim.x + threadIdx.x;
  int cols = 28 * fc;
  if (idx >= cols * finp) return;
  int j = idx / finp, k = idx % finp;
  float v = 0.f;
  if (k < fin) {
    if (j < SKNOTS * fc) {
      int s = j / fc, o = o0 + (j % fc);
      v = W[(s * fin + k) * fout + o];
    } else {
      int o = o0 + (j - SKNOTS * fc);
      v = root[k * fout + o];
    }
  }
  WtT[idx] = (_Float16)v;
}

// ---------------- WMMA GEMM: Y[N,cols] = Xh[N,finp] @ WtT[cols,finp]^T ----------------
// 4 waves/block, one 16x16 C tile per wave (4 row-tiles, shared 16-col B strip).
// Whole K-strips DMA'd to LDS via async LDS-DMA (ASYNCcnt), then a barrier-free
// K-loop of ds_load_b128 + v_wmma. Fragment layouts per ISA 7.12.2 (wave32).
__global__ void __launch_bounds__(128)
k_gemm_wmma(const _Float16* __restrict__ Xh, const _Float16* __restrict__ WtT,
            float* __restrict__ Y, int finp, int cols) {
  __shared__ __align__(16) _Float16 As[4][16 * FINP_MAX];   // per-wave A strip
  __shared__ __align__(16) _Float16 Bs[16 * FINP_MAX];      // shared B strip
  const int tid = threadIdx.x, wave = tid >> 5, lane = tid & 31;
  int rowtile = blockIdx.y * 4 + wave;
  if (rowtile > NROWT - 1) rowtile = NROWT - 1;   // duplicate tail work (bit-identical)
  const int row0 = rowtile * 16;
  const int col0 = blockIdx.x * 16;
  const int strip = 32 * finp;                    // bytes per 16-row/col strip (mult of 1024)

  {  // async DMA: this wave's full A strip (contiguous rows of Xh)
    unsigned ldsA = (unsigned)(unsigned long long)&As[wave][0];
    const char* gA = (const char*)(Xh + (size_t)row0 * finp);
    for (int off = lane * 16; off < strip; off += 512) {
      unsigned l = ldsA + off;
      const char* g = gA + off;
      asm volatile("global_load_async_to_lds_b128 %0, %1, off"
                   :: "v"(l), "v"(g) : "memory");
    }
  }
  if (wave == 0) {  // async DMA: full B strip (contiguous K-major cols of WtT)
    unsigned ldsB = (unsigned)(unsigned long long)&Bs[0];
    const char* gB = (const char*)(WtT + (size_t)col0 * finp);
    for (int off = lane * 16; off < strip; off += 512) {
      unsigned l = ldsB + off;
      const char* g = gB + off;
      asm volatile("global_load_async_to_lds_b128 %0, %1, off"
                   :: "v"(l), "v"(g) : "memory");
    }
  }
  asm volatile("s_wait_asynccnt 0x0" ::: "memory");
  __syncthreads();

  const int M = lane & 15, hh = lane >> 4;
  const _Float16* aRow = &As[wave][M * finp];     // A row for this lane
  const _Float16* bCol = &Bs[M * finp];           // B column for this lane
  v8f c = {};
  const int ksteps = finp >> 5;
  union FragU { float4 f4[2]; v16h v; };
  for (int t = 0; t < ksteps; ++t) {
    const int k0 = t * 32;
    FragU ua, ub;
    // A 16x32 layout: VGPR j holds K = (j<4 ? 2j : 16+2(j-4)) + 8*half
    //   -> two contiguous 16B spans: [k0+8h, +8) and [k0+16+8h, +8)
    ua.f4[0] = *(const float4*)(aRow + k0 + 8 * hh);
    ua.f4[1] = *(const float4*)(aRow + k0 + 16 + 8 * hh);
    // B 32x16 layout: same K order as A ; column = lane&15
    //   -> two contiguous 16B spans: [k0+8h, +8) and [k0+16+8h, +8)
    ub.f4[0] = *(const float4*)(bCol + k0 + 8 * hh);
    ub.f4[1] = *(const float4*)(bCol + k0 + 16 + 8 * hh);
    c = __builtin_amdgcn_wmma_f32_16x16x32_f16(false, ua.v, false, ub.v,
                                               (short)0, c, false, false);
  }
  const int nn = lane & 15;
#pragma unroll
  for (int r = 0; r < 8; ++r) {                   // C layout: M = r + 8*half
    Y[(row0 + r + 8 * hh) * cols + col0 + nn] = c[r];
  }
}

// ---------------- scatter: tmp[dst, o0+.] += bas * Y[src, wi*fc+.] ----------------
__global__ void k_scatter(const float* __restrict__ ebas, const int* __restrict__ ewi,
                          const int* __restrict__ ei, const float* __restrict__ Y,
                          float* __restrict__ tmp, int fout, int fc, int o0, int cols) {
  int e8 = blockIdx.x * blockDim.x + threadIdx.x;
  if (e8 >= NEDGES * 8) return;
  int e = e8 >> 3;
  float bas = ebas[e8];
  int wi = ewi[e8];
  int src = ei[e], dst = ei[NEDGES + e];
  const float* yp = &Y[src * cols + wi * fc];
  float* tp = &tmp[dst * fout + o0];
  for (int o = 0; o < fc; o += 4) {
    float4 y = *(const float4*)&yp[o];
    atomicAdd(&tp[o + 0], bas * y.x);
    atomicAdd(&tp[o + 1], bas * y.y);
    atomicAdd(&tp[o + 2], bas * y.z);
    atomicAdd(&tp[o + 3], bas * y.w);
  }
}

// out = tmp*deg_inv + (x@root from Y slot 27) + bias  [+ ELU]
__global__ void k_finalize(const float* __restrict__ tmp, const float* __restrict__ Y,
                           const float* __restrict__ deg_inv, const float* __restrict__ bias,
                           float* __restrict__ out, int fout, int fc, int o0, int cols, int doElu) {
  int idx = blockIdx.x * blockDim.x + threadIdx.x;
  if (idx >= NNODES * fc) return;
  int n = idx / fc, o = idx % fc;
  float v = tmp[n * fout + o0 + o] * deg_inv[n] + Y[n * cols + SKNOTS * fc + o] + bias[o0 + o];
  if (doElu) v = (v > 0.f) ? v : expm1f(v);
  out[n * fout + o0 + o] = v;
}

// ---------------- batchnorm (biased var) + ELU ----------------
__global__ void k_bn_stats(const float* __restrict__ h, float* __restrict__ stats, int C) {
  int c = blockIdx.x;
  __shared__ float ss[256], ss2[256];
  float s = 0.f, s2 = 0.f;
  for (int n = threadIdx.x; n < NNODES; n += 256) {
    float v = h[n * C + c];
    s += v; s2 += v * v;
  }
  ss[threadIdx.x] = s; ss2[threadIdx.x] = s2;
  __syncthreads();
  for (int st = 128; st > 0; st >>= 1) {
    if (threadIdx.x < st) { ss[threadIdx.x] += ss[threadIdx.x + st]; ss2[threadIdx.x] += ss2[threadIdx.x + st]; }
    __syncthreads();
  }
  if (threadIdx.x == 0) {
    float mu = ss[0] / (float)NNODES;
    float var = ss2[0] / (float)NNODES - mu * mu;
    stats[c] = mu;
    stats[C + c] = rsqrtf(var + BN_EPS);
  }
}

__global__ void k_bn_apply(float* __restrict__ h, const float* __restrict__ stats,
                           const float* __restrict__ gamma, const float* __restrict__ beta, int C) {
  int idx = blockIdx.x * blockDim.x + threadIdx.x;
  if (idx >= NNODES * C) return;
  int c = idx % C;
  float v = gamma[c] * (h[idx] - stats[c]) * stats[C + c] + beta[c];
  h[idx] = (v > 0.f) ? v : expm1f(v);
}

// concat helper: dst[n, off:off+srcW] = src[n, :]
__global__ void k_copy_cols(float* __restrict__ dst, int dstW, int off,
                            const float* __restrict__ src, int srcW) {
  int idx = blockIdx.x * blockDim.x + threadIdx.x;
  if (idx >= NNODES * srcW) return;
  int n = idx / srcW, c = idx % srcW;
  dst[n * dstW + off + c] = src[n * srcW + c];
}

// ---------------- host orchestration ----------------

struct Conv { const float* W; const float* root; const float* bias; };

struct Ws {
  float* ebas; int* ewi; float* deg;
  float* t1; float* t2; float* c3in; float* blkout; float* r4; float* cur;
  _Float16* Xh; _Float16* WtT; float* Y; float* tmp; float* stats;
  const int* ei;
};

static void run_conv(const float* X, int fin, Conv cv, int fout, float* out, bool elu,
                     const Ws& w, hipStream_t s) {
  const int finp = cdiv(fin, 32) * 32;
  k_pack_x<<<cdiv(NNODES * finp, 256), 256, 0, s>>>(X, w.Xh, fin, finp);
  k_zero<<<cdiv(NNODES * fout, 256), 256, 0, s>>>(w.tmp, NNODES * fout);
  int fc = fout < 32 ? fout : 32;          // fout always a multiple of fc; fc % 4 == 0
  for (int o0 = 0; o0 < fout; o0 += fc) {
    int cols = 28 * fc;
    k_pack_w<<<cdiv(cols * finp, 256), 256, 0, s>>>(cv.W, cv.root, w.WtT, fin, finp, fout, fc, o0);
    dim3 g(cols / 16, cdiv(NROWT, 4));
    k_gemm_wmma<<<g, 128, 0, s>>>(w.Xh, w.WtT, w.Y, finp, cols);
    k_scatter<<<cdiv(NEDGES * 8, 256), 256, 0, s>>>(w.ebas, w.ewi, w.ei, w.Y, w.tmp,
                                                    fout, fc, o0, cols);
    k_finalize<<<cdiv(NNODES * fc, 256), 256, 0, s>>>(w.tmp, w.Y, w.deg, cv.bias, out,
                                                      fout, fc, o0, cols, elu ? 1 : 0);
  }
}

static void run_block(const float* res, int fin, int mid, int fout,
                      void* const* d_in, int pbase, const float* pos, float* out,
                      const Ws& w, hipStream_t s) {
  Conv c1{(const float*)d_in[pbase + 0], (const float*)d_in[pbase + 1], (const float*)d_in[pbase + 2]};
  Conv c2{(const float*)d_in[pbase + 3], (const float*)d_in[pbase + 4], (const float*)d_in[pbase + 5]};
  Conv c3{(const float*)d_in[pbase + 6], (const float*)d_in[pbase + 7], (const float*)d_in[pbase + 8]};
  const float* gamma = (const float*)d_in[pbase + 9];
  const float* beta  = (const float*)d_in[pbase + 10];

  run_conv(res, fin, c1, mid, w.t1, /*elu=*/true, w, s);
  int C = 2 * mid;
  run_conv(w.t1, mid, c2, C, w.t2, /*elu=*/false, w, s);
  k_bn_stats<<<C, 256, 0, s>>>(w.t2, w.stats, C);
  k_bn_apply<<<cdiv(NNODES * C, 256), 256, 0, s>>>(w.t2, w.stats, gamma, beta, C);
  int w3 = C + 3;
  k_copy_cols<<<cdiv(NNODES * C, 256), 256, 0, s>>>(w.c3in, w3, 0, w.t2, C);
  k_copy_cols<<<cdiv(NNODES * 3, 256), 256, 0, s>>>(w.c3in, w3, C, pos, 3);
  run_conv(w.c3in, w3, c3, fout, out, /*elu=*/false, w, s);
}

extern "C" void kernel_launch(void* const* d_in, const int* in_sizes, int n_in,
                              void* d_out, int out_size, void* d_ws, size_t ws_size,
                              hipStream_t stream) {
  const float* x   = (const float*)d_in[0];
  const float* pos = (const float*)d_in[1];
  const float* ea  = (const float*)d_in[2];
  const int*   ei  = (const int*)d_in[3];

  // workspace carve-out (256B aligned)
  char* base = (char*)d_ws;
  size_t off = 0;
  auto take = [&](size_t bytes) { char* p = base + off; off += (bytes + 255) & ~(size_t)255; return p; };
  Ws w;
  w.ebas  = (float*)take((size_t)NEDGES * 8 * 4);
  w.ewi   = (int*)  take((size_t)NEDGES * 8 * 4);
  w.deg   = (float*)take((size_t)NNODES * 4);
  w.t1    = (float*)take((size_t)NNODES * 128 * 4);
  w.t2    = (float*)take((size_t)NNODES * 256 * 4);
  w.c3in  = (float*)take((size_t)NNODES * 260 * 4);
  w.blkout= (float*)take((size_t)NNODES * 64 * 4);
  w.r4    = (float*)take((size_t)NNODES * 8 * 4);
  w.cur   = (float*)take((size_t)NNODES * 80 * 4);
  w.Xh    = (_Float16*)take((size_t)NNODES * FINP_MAX * 2);
  w.WtT   = (_Float16*)take((size_t)28 * 32 * FINP_MAX * 2);
  w.Y     = (float*)take((size_t)NNODES * 28 * 32 * 4);
  w.tmp   = (float*)take((size_t)NNODES * 256 * 4);
  w.stats = (float*)take((size_t)2 * 256 * 4);
  w.ei    = ei;
  hipStream_t s = stream;

  // edge basis + degrees (reused by every conv)
  k_edge_basis<<<cdiv(NEDGES, 256), 256, 0, s>>>(ea, w.ebas, w.ewi);
  k_zero<<<cdiv(NNODES, 256), 256, 0, s>>>(w.deg, NNODES);
  k_deg<<<cdiv(NEDGES, 256), 256, 0, s>>>(ei, w.deg);
  k_deg_inv<<<cdiv(NNODES, 256), 256, 0, s>>>(w.deg);

  auto P = [&](int blk) { return 4 + blk * 11; };

  // r = block0(x)          3 -> 16
  k_copy_cols<<<cdiv(NNODES * 3, 256), 256, 0, s>>>(w.cur, 3, 0, x, 3);
  run_block(w.cur, 3, 8, 16, d_in, P(0), pos, w.blkout, w, s);
  k_copy_cols<<<cdiv(NNODES * 16, 256), 256, 0, s>>>(w.cur, 16, 0, w.blkout, 16);
  // block1: 16 -> 64
  run_block(w.cur, 16, 32, 64, d_in, P(1), pos, w.blkout, w, s);
  k_copy_cols<<<cdiv(NNODES * 64, 256), 256, 0, s>>>(w.cur, 64, 0, w.blkout, 64);
  // block2: 64 -> 64
  run_block(w.cur, 64, 128, 64, d_in, P(2), pos, w.blkout, w, s);
  k_copy_cols<<<cdiv(NNODES * 64, 256), 256, 0, s>>>(w.cur, 64, 0, w.blkout, 64);
  // block3: 64 -> 8   (r4)
  run_block(w.cur, 64, 16, 8, d_in, P(3), pos, w.blkout, w, s);
  k_copy_cols<<<cdiv(NNODES * 8, 256), 256, 0, s>>>(w.r4, 8, 0, w.blkout, 8);
  // concat(r4, x) -> 11
  k_copy_cols<<<cdiv(NNODES * 8, 256), 256, 0, s>>>(w.cur, 11, 0, w.r4, 8);
  k_copy_cols<<<cdiv(NNODES * 3, 256), 256, 0, s>>>(w.cur, 11, 8, x, 3);
  // block4: 11 -> 32
  run_block(w.cur, 11, 16, 32, d_in, P(4), pos, w.blkout, w, s);
  k_copy_cols<<<cdiv(NNODES * 32, 256), 256, 0, s>>>(w.cur, 32, 0, w.blkout, 32);
  // block5: 32 -> 64
  run_block(w.cur, 32, 32, 64, d_in, P(5), pos, w.blkout, w, s);
  k_copy_cols<<<cdiv(NNODES * 64, 256), 256, 0, s>>>(w.cur, 64, 0, w.blkout, 64);
  // block6: 64 -> 64
  run_block(w.cur, 64, 128, 64, d_in, P(6), pos, w.blkout, w, s);
  // concat(r, r4, x) -> 75
  k_copy_cols<<<cdiv(NNODES * 64, 256), 256, 0, s>>>(w.cur, 75, 0, w.blkout, 64);
  k_copy_cols<<<cdiv(NNODES * 8, 256), 256, 0, s>>>(w.cur, 75, 64, w.r4, 8);
  k_copy_cols<<<cdiv(NNODES * 3, 256), 256, 0, s>>>(w.cur, 75, 72, x, 3);
  // block7: 75 -> 4  (final output)
  run_block(w.cur, 75, 16, 4, d_in, P(7), pos, (float*)d_out, w, s);
}
